// SelectiveSSM_21500606284031
// MI455X (gfx1250) — compile-verified, matrix operand order checked by reading, not run on hardware
//
#include <hip/hip_runtime.h>
#include <hip/hip_bf16.h>
#include <math.h>

// ---------------- CDNA5 WMMA types ----------------
typedef __attribute__((ext_vector_type(16))) __bf16 v16bf;
typedef __attribute__((ext_vector_type(8)))  float  v8f;

// Model dims (compile-time constants from the reference)
#define BATCH   2
#define SEQ     2048
#define DMODEL  1024
#define DINNER  2048
#define DSTATE  16
#define DCONV   4
#define DTRANK  64
#define NPROJ   (DTRANK + 2 * DSTATE)   // 96
#define MROWS   (BATCH * SEQ)           // 4096

// fp32 -> bf16, round-to-nearest-even
__device__ __forceinline__ __bf16 f2bf(float f) {
    unsigned u = __builtin_bit_cast(unsigned, f);
    u += 0x7fffu + ((u >> 16) & 1u);
    unsigned short h = (unsigned short)(u >> 16);
    return __builtin_bit_cast(__bf16, h);
}

__device__ __forceinline__ float fast_sigmoid(float v) {
    return 1.f / (1.f + __expf(-v));            // v_exp_f32 (TRANS unit)
}
__device__ __forceinline__ float fast_softplus(float v) {
    return (v > 20.f) ? v : __logf(1.f + __expf(v));
}

union Frag16 {                 // one 16x32(bf16) WMMA operand: 32 bytes/lane
    v16bf v;
    uint4 q[2];
};

// ---------------------------------------------------------------------------
// bf16 WMMA GEMM: C[M x N] = A[M x K] * B[K x N], with A bf16 row-major
// (leading dim lda) and B given pre-transposed bf16 BT[N x K] (leading dim
// ldb==K). Each wave computes a (16*MT) x (16*NT) C tile.
// Fragment layout (ISA 7.12.2, wave32): lane&15 = row(A)/col(B);
// K halves: i<8 -> k0+kh+i, i>=8 -> k0+16+kh+(i-8), kh=(lane>>4)*8.
// => per fragment: two aligned 16-byte loads at k0+kh and k0+16+kh.
// Prefetch only the A (activation) stream: BT weight tiles are reused by all
// M-tiles and stay L2-resident (192 MB); prefetching them only adds VGPR
// pressure (caused scratch spills in the hot loop last round).
// EPI: 0 = plain fp32 store; 1 = softplus(x + bias[col]).
// If Cbf != nullptr, additionally store bf16 copy of C.
// ---------------------------------------------------------------------------
template<int MT, int NT, int EPI>
__global__ void wmma_gemm_bf16(const __bf16* __restrict__ A,
                               const __bf16* __restrict__ BT,
                               float* __restrict__ C,
                               __bf16* __restrict__ Cbf,
                               const float* __restrict__ bias,
                               int M, int N, int K, int lda, int ldb, int ldc)
{
    const int lane = threadIdx.x & 31;
    const int wave = threadIdx.x >> 5;
    const int wpb  = blockDim.x >> 5;
    const int tilesN = N / (16 * NT);
    const int tilesM = M / (16 * MT);
    const int tile   = blockIdx.x * wpb + wave;
    if (tile >= tilesM * tilesN) return;      // wave-uniform exit (EXEC all-1s for WMMA)
    const int tm = (tile / tilesN) * (16 * MT);
    const int tn = (tile % tilesN) * (16 * NT);

    const int r  = lane & 15;
    const int kh = (lane >> 4) * 8;

    v8f zero;
    #pragma unroll
    for (int i = 0; i < 8; ++i) zero[i] = 0.f;
    v8f acc[MT][NT];
    #pragma unroll
    for (int m = 0; m < MT; ++m)
        #pragma unroll
        for (int j = 0; j < NT; ++j) acc[m][j] = zero;

    #pragma unroll 2
    for (int k0 = 0; k0 < K; k0 += 32) {
        Frag16 a[MT];
        #pragma unroll
        for (int m = 0; m < MT; ++m) {
            const __bf16* Ab = A + (size_t)(tm + m * 16 + r) * lda + k0;
            __builtin_prefetch(Ab + 64, 0, 0);          // K-tile two ahead (k0+64) -> global_prefetch_b8
            a[m].q[0] = *(const uint4*)(Ab + kh);
            a[m].q[1] = *(const uint4*)(Ab + 16 + kh);
        }
        #pragma unroll
        for (int j = 0; j < NT; ++j) {
            Frag16 b;
            const __bf16* Bb = BT + (size_t)(tn + j * 16 + r) * ldb + k0;
            b.q[0] = *(const uint4*)(Bb + kh);
            b.q[1] = *(const uint4*)(Bb + 16 + kh);
            #pragma unroll
            for (int m = 0; m < MT; ++m)
                acc[m][j] = __builtin_amdgcn_wmma_f32_16x16x32_bf16(
                                false, a[m].v, false, b.v, (short)0, acc[m][j], false, false);
        }
    }

    const int col = lane & 15;
    #pragma unroll
    for (int m = 0; m < MT; ++m) {
        const int rowBase = tm + m * 16 + ((lane >> 4) ? 8 : 0);
        #pragma unroll
        for (int j = 0; j < NT; ++j) {
            const int c = tn + j * 16 + col;
            #pragma unroll
            for (int rr = 0; rr < 8; ++rr) {
                float v = acc[m][j][rr];
                if (EPI == 1) {                       // softplus(x + b_dt)
                    v = fast_softplus(v + bias[c]);
                }
                const size_t idx = (size_t)(rowBase + rr) * ldc + c;
                C[idx] = v;
                if (Cbf) Cbf[idx] = f2bf(v);
            }
        }
    }
}

// ---------------------------------------------------------------------------
// fp32 -> bf16 elementwise pack
// ---------------------------------------------------------------------------
__global__ void pack_bf16_kernel(const float* __restrict__ src,
                                 __bf16* __restrict__ dst, long n)
{
    long i = (long)blockIdx.x * blockDim.x + threadIdx.x;
    if (i < n) dst[i] = f2bf(src[i]);
}

// ---------------------------------------------------------------------------
// fp32 [R x C] -> bf16 transposed [C x R], LDS-tiled (32x32, pad 33)
// ---------------------------------------------------------------------------
__global__ void transpose_pack_bf16(const float* __restrict__ src,
                                    __bf16* __restrict__ dst, int R, int C)
{
    __shared__ __bf16 tile[32][33];
    const int c0 = blockIdx.x * 32, r0 = blockIdx.y * 32;
    const int tx = threadIdx.x, ty = threadIdx.y;      // block (32, 8)
    #pragma unroll
    for (int i = 0; i < 32; i += 8) {
        const int rr = r0 + ty + i, cc = c0 + tx;
        if (rr < R && cc < C) tile[ty + i][tx] = f2bf(src[(size_t)rr * C + cc]);
    }
    __syncthreads();
    #pragma unroll
    for (int i = 0; i < 32; i += 8) {
        const int rr = c0 + ty + i, cc = r0 + tx;      // dst is [C x R]
        if (rr < C && cc < R) dst[(size_t)rr * R + cc] = tile[tx][ty + i];
    }
}

// ---------------------------------------------------------------------------
// Depthwise causal conv1d (K=4, left pad 3) + bias + SiLU on the x-branch.
// xz: [B*L, 2*DINNER] fp32; writes xc fp32 (for scan) and bf16 (for GEMM).
// ---------------------------------------------------------------------------
__global__ void conv_silu_kernel(const float* __restrict__ xz,
                                 const float* __restrict__ conv_w,
                                 const float* __restrict__ conv_b,
                                 float* __restrict__ xc,
                                 __bf16* __restrict__ xc_bf)
{
    const int idx = blockIdx.x * blockDim.x + threadIdx.x;
    if (idx >= BATCH * SEQ * DINNER) return;
    const int d = idx % DINNER;
    const int l = (idx / DINNER) % SEQ;
    const int b = idx / (DINNER * SEQ);

    float acc = conv_b[d];
    #pragma unroll
    for (int j = 0; j < DCONV; ++j) {
        const int t = l + j - (DCONV - 1);
        if (t >= 0)
            acc += conv_w[d * DCONV + j] *
                   xz[(size_t)(b * SEQ + t) * (2 * DINNER) + d];
    }
    const float s = acc * fast_sigmoid(acc);            // SiLU
    const size_t o = (size_t)(b * SEQ + l) * DINNER + d;
    xc[o]    = s;
    xc_bf[o] = f2bf(s);
}

// ---------------------------------------------------------------------------
// Selective scan: one thread per (batch, channel); 16-wide state in registers.
// h = exp(dt*A)*h + dt*B*xc ;  y = <h,C> + xc*D ;  y *= silu(z) ; store bf16.
// exp via v_exp_f32 (TRANS unit) to shorten the 2048-step dependent chain.
// ---------------------------------------------------------------------------
__global__ void scan_kernel(const float* __restrict__ xc,
                            const float* __restrict__ proj,   // [B*L, 96] fp32
                            const float* __restrict__ dt,     // [B*L, DINNER]
                            const float* __restrict__ A_log,  // [DINNER, 16]
                            const float* __restrict__ D_skip, // [DINNER]
                            const float* __restrict__ xz,     // z = xz[:, DINNER:]
                            __bf16* __restrict__ y)           // [B*L, DINNER] bf16
{
    const int ch = blockIdx.x * blockDim.x + threadIdx.x;
    if (ch >= BATCH * DINNER) return;
    const int d = ch % DINNER;
    const int b = ch / DINNER;

    float Arow[DSTATE], h[DSTATE];
    #pragma unroll
    for (int n = 0; n < DSTATE; ++n) {
        Arow[n] = -__expf(A_log[d * DSTATE + n]);
        h[n] = 0.f;
    }
    const float Dd = D_skip[d];

    for (int t = 0; t < SEQ; ++t) {
        const size_t rm  = (size_t)(b * SEQ + t);
        const float dtv  = dt[rm * DINNER + d];
        const float xcv  = xc[rm * DINNER + d];
        const float* Bp  = proj + rm * NPROJ + DTRANK;
        const float* Cp  = Bp + DSTATE;
        const float dx   = dtv * xcv;
        float yv = 0.f;
        #pragma unroll
        for (int n = 0; n < DSTATE; ++n) {
            const float dA = __expf(dtv * Arow[n]);     // v_exp_f32
            h[n] = dA * h[n] + dx * Bp[n];
            yv  += h[n] * Cp[n];
        }
        yv += xcv * Dd;
        const float z = xz[rm * (2 * DINNER) + DINNER + d];
        yv *= z * fast_sigmoid(z);
        y[rm * DINNER + d] = f2bf(yv);
    }
}

// ---------------------------------------------------------------------------
extern "C" void kernel_launch(void* const* d_in, const int* in_sizes, int n_in,
                              void* d_out, int out_size, void* d_ws, size_t ws_size,
                              hipStream_t stream) {
    const float* x      = (const float*)d_in[0];
    const float* W_in   = (const float*)d_in[1];   // [1024, 4096]
    const float* conv_w = (const float*)d_in[2];
    const float* conv_b = (const float*)d_in[3];
    const float* W_xprj = (const float*)d_in[4];   // [2048, 96]
    const float* W_dt   = (const float*)d_in[5];   // [64, 2048]
    const float* b_dt   = (const float*)d_in[6];
    const float* A_log  = (const float*)d_in[7];
    const float* D_skip = (const float*)d_in[8];
    const float* W_out  = (const float*)d_in[9];   // [2048, 1024]
    float* out = (float*)d_out;

    // ---- workspace carve-up (all 256B aligned) ----
    char* p = (char*)d_ws;
    auto alloc = [&](size_t bytes) { char* q = p; p += (bytes + 255) & ~(size_t)255; return q; };
    float*  xz      = (float*) alloc((size_t)MROWS * 2 * DINNER * 4);   // 64 MB
    float*  xc      = (float*) alloc((size_t)MROWS * DINNER * 4);       // 32 MB
    float*  proj    = (float*) alloc((size_t)MROWS * NPROJ * 4);        // 1.5 MB
    float*  dt      = (float*) alloc((size_t)MROWS * DINNER * 4);       // 32 MB
    __bf16* x_bf    = (__bf16*)alloc((size_t)MROWS * DMODEL * 2);       // 8 MB
    __bf16* Win_t   = (__bf16*)alloc((size_t)2 * DINNER * DMODEL * 2);  // 8 MB  [4096 x 1024]
    __bf16* xc_bf   = (__bf16*)alloc((size_t)MROWS * DINNER * 2);       // 16 MB
    __bf16* Wxp_t   = (__bf16*)alloc((size_t)NPROJ * DINNER * 2);       // [96 x 2048]
    __bf16* proj_bf = (__bf16*)alloc((size_t)MROWS * NPROJ * 2);        // [4096 x 96]
    __bf16* Wdt_t   = (__bf16*)alloc((size_t)DINNER * DTRANK * 2);      // [2048 x 64]
    __bf16* y_bf    = (__bf16*)alloc((size_t)MROWS * DINNER * 2);       // 16 MB
    __bf16* Wout_t  = (__bf16*)alloc((size_t)DMODEL * DINNER * 2);      // [1024 x 2048]

    const int BLK = 256, WPB = BLK / 32;
    auto blocksFor = [](long n, int per) { return (int)((n + per - 1) / per); };

    // 0) packs / transposes to bf16
    pack_bf16_kernel<<<blocksFor((long)MROWS * DMODEL, BLK), BLK, 0, stream>>>(
        x, x_bf, (long)MROWS * DMODEL);
    transpose_pack_bf16<<<dim3((2 * DINNER) / 32, DMODEL / 32), dim3(32, 8), 0, stream>>>(
        W_in, Win_t, DMODEL, 2 * DINNER);
    transpose_pack_bf16<<<dim3(NPROJ / 32, DINNER / 32), dim3(32, 8), 0, stream>>>(
        W_xprj, Wxp_t, DINNER, NPROJ);
    transpose_pack_bf16<<<dim3(DINNER / 32, DTRANK / 32), dim3(32, 8), 0, stream>>>(
        W_dt, Wdt_t, DTRANK, DINNER);
    transpose_pack_bf16<<<dim3(DMODEL / 32, DINNER / 32), dim3(32, 8), 0, stream>>>(
        W_out, Wout_t, DINNER, DMODEL);

    // 1) xz = x @ W_in            M=4096 K=1024 N=4096   (32x64 tile / wave)
    {
        int tiles = (MROWS / 32) * ((2 * DINNER) / 64);
        wmma_gemm_bf16<2, 4, 0><<<blocksFor(tiles, WPB), BLK, 0, stream>>>(
            x_bf, Win_t, xz, nullptr, nullptr,
            MROWS, 2 * DINNER, DMODEL, DMODEL, DMODEL, 2 * DINNER);
    }
    // 2) xc = silu(depthwise_conv(x_in) + conv_b)   (fp32 + bf16)
    conv_silu_kernel<<<blocksFor((long)BATCH * SEQ * DINNER, BLK), BLK, 0, stream>>>(
        xz, conv_w, conv_b, xc, xc_bf);
    // 3) proj = xc @ W_xproj      M=4096 K=2048 N=96   (dual-store fp32 + bf16)
    {
        int tiles = (MROWS / 32) * (NPROJ / 32);
        wmma_gemm_bf16<2, 2, 0><<<blocksFor(tiles, WPB), BLK, 0, stream>>>(
            xc_bf, Wxp_t, proj, proj_bf, nullptr,
            MROWS, NPROJ, DINNER, DINNER, DINNER, NPROJ);
    }
    // 4) dt = softplus(proj[:, :64] @ W_dt + b_dt)   M=4096 K=64 N=2048
    {
        int tiles = (MROWS / 32) * (DINNER / 64);
        wmma_gemm_bf16<2, 4, 1><<<blocksFor(tiles, WPB), BLK, 0, stream>>>(
            proj_bf, Wdt_t, dt, nullptr, b_dt,
            MROWS, DINNER, DTRANK, NPROJ, DTRANK, DINNER);
    }
    // 5) selective scan + skip + gate  -> y (bf16)
    scan_kernel<<<blocksFor((long)BATCH * DINNER, BLK), BLK, 0, stream>>>(
        xc, proj, dt, A_log, D_skip, xz, y_bf);
    // 6) out = y @ W_out          M=4096 K=2048 N=1024
    {
        int tiles = (MROWS / 32) * (DMODEL / 64);
        wmma_gemm_bf16<2, 4, 0><<<blocksFor(tiles, WPB), BLK, 0, stream>>>(
            y_bf, Wout_t, out, nullptr, nullptr,
            MROWS, DMODEL, DINNER, DINNER, DINNER, DMODEL);
    }
}
